// Eagle3Attention_85194971283568
// MI455X (gfx1250) — matrix operand order repaired and checked by reading, not compiled, not run on hardware
//
#include <hip/hip_runtime.h>
#include <stdint.h>

// ---------------- problem constants ----------------
#define T_LEN  2048
#define HIDDEN 4096
#define NHQ    32          // query heads
#define NKV    8           // kv heads
#define DH     128         // head dim
#define GRP    (NHQ/NKV)   // 4
#define K_IN   (2*HIDDEN)  // 8192
#define VCOLS  (NKV*DH)    // 1024

typedef __attribute__((ext_vector_type(16))) __bf16 v16bf;
typedef __attribute__((ext_vector_type(8)))  __bf16 v8bf;
typedef __attribute__((ext_vector_type(8)))  float  v8f;

// ---------------- helpers ----------------
static __device__ __forceinline__ __bf16 f2bf(float f) {
    uint32_t u = __builtin_bit_cast(uint32_t, f);
    uint32_t r = (u + 0x7FFFu + ((u >> 16) & 1u)) >> 16;  // round-to-nearest-even
    return __builtin_bit_cast(__bf16, (uint16_t)r);
}
static __device__ __forceinline__ uint32_t pack2bf(float lo, float hi) {
    uint32_t l = __builtin_bit_cast(uint16_t, f2bf(lo));
    uint32_t h = __builtin_bit_cast(uint16_t, f2bf(hi));
    return l | (h << 16);
}

static __device__ __forceinline__ v16bf combine16(v8bf lo, v8bf hi) {
    return __builtin_shufflevector(lo, hi, 0,1,2,3,4,5,6,7,8,9,10,11,12,13,14,15);
}
// A-fragment: two contiguous 16B chunks (K..K+7 and K+16..K+23 of one row)
static __device__ __forceinline__ v16bf load_a_frag(const __bf16* __restrict__ p) {
    return combine16(*(const v8bf*)p, *(const v8bf*)(p + 16));
}

static __device__ __forceinline__ v8f wmma_bf16(v16bf a, v16bf b, v8f c) {
    return __builtin_amdgcn_wmma_f32_16x16x32_bf16(false, a, false, b, (short)0, c, false, false);
}

// ---------------- f32 -> bf16 convert, 8 elements/thread ----------------
__global__ void f32_to_bf16_vec8(const float* __restrict__ in, __bf16* __restrict__ out, long n8) {
    long i = (long)blockIdx.x * blockDim.x + threadIdx.x;   // index in units of 8 elements
    if (i >= n8) return;
    const float4* p = (const float4*)(in + i * 8);
    float4 a = p[0];
    float4 b = p[1];
    uint4 o;
    o.x = pack2bf(a.x, a.y);
    o.y = pack2bf(a.z, a.w);
    o.z = pack2bf(b.x, b.y);
    o.w = pack2bf(b.z, b.w);
    *(uint4*)(out + i * 8) = o;
}

// ---------------- generic GEMM: C[M,N] = A[M,K](bf16) * W[N,K]^T(bf16), C f32 ----------------
// block = 256 thr = 8 waves; wave tile 64x64 (4 Mfrag x 4 Nfrag); block tile 128x256
__global__ __launch_bounds__(256)
void gemm_bf16_wmma(const __bf16* __restrict__ A, const __bf16* __restrict__ B,
                    float* __restrict__ C, int M, int N, int K) {
    const int lane   = threadIdx.x & 31;
    const int wave   = threadIdx.x >> 5;
    const int l16    = lane & 15;
    const int hiHalf = lane >> 4;                 // 0: lanes 0-15, 1: lanes 16-31
    const int mBase  = blockIdx.y * 128 + (wave >> 2) * 64;
    const int nBase  = blockIdx.x * 256 + (wave & 3) * 64;

    const v8f zero = {0.f,0.f,0.f,0.f,0.f,0.f,0.f,0.f};
    v8f acc[4][4];
#pragma unroll
    for (int im = 0; im < 4; ++im)
#pragma unroll
        for (int jn = 0; jn < 4; ++jn) acc[im][jn] = zero;

    const int a_off = hiHalf * 8;    // A frag: lanes>=16 start at K+8
    const int b_off = hiHalf * 8;    // B frag: lanes>=16 start at K+8

    const __bf16* Arow[4];
    const __bf16* Brow[4];
#pragma unroll
    for (int im = 0; im < 4; ++im)
        Arow[im] = A + (size_t)(mBase + im * 16 + l16) * K + a_off;
#pragma unroll
    for (int jn = 0; jn < 4; ++jn)
        Brow[jn] = B + (size_t)(nBase + jn * 16 + l16) * K + b_off;

    for (int k0 = 0; k0 < K; k0 += 32) {
        v16bf af[4];
#pragma unroll
        for (int im = 0; im < 4; ++im)
            af[im] = load_a_frag(Arow[im] + k0);
#pragma unroll
        for (int jn = 0; jn < 4; ++jn) {
            v16bf bf = load_a_frag(Brow[jn] + k0);  // two 8-elem chunks of W row (K..K+7, K+16..K+23)
#pragma unroll
            for (int im = 0; im < 4; ++im)
                acc[im][jn] = wmma_bf16(af[im], bf, acc[im][jn]);
        }
    }
#pragma unroll
    for (int im = 0; im < 4; ++im)
#pragma unroll
        for (int jn = 0; jn < 4; ++jn) {
            int n = nBase + jn * 16 + l16;
#pragma unroll
            for (int r = 0; r < 8; ++r) {
                int m = mBase + im * 16 + r + hiHalf * 8;       // C layout: VGPR r -> row r / r+8
                C[(size_t)m * N + n] = acc[im][jn][r];
            }
        }
}

// ---------------- RoPE: f32 (T, nh*DH) -> bf16, rotate-half ----------------
__global__ void rope_f32_to_bf16(const float* __restrict__ src, __bf16* __restrict__ dst,
                                 const int* __restrict__ positions, int nheads) {
    int bid = blockIdx.x;                 // t * nheads + head
    int t = bid / nheads;
    int hh = bid - t * nheads;
    int i = threadIdx.x;                  // 0..63
    const size_t base = ((size_t)t * nheads + hh) * DH;
    float inv = powf(10000.0f, -(2.0f * (float)i) / (float)DH);
    float ang = (float)positions[t] * inv;
    float c = cosf(ang), s = sinf(ang);
    float t1 = src[base + i];
    float t2 = src[base + i + 64];
    dst[base + i]      = f2bf(t1 * c - t2 * s);
    dst[base + i + 64] = f2bf(t2 * c + t1 * s);
}

// ---------------- V transpose: f32 (T, VCOLS) -> bf16 (VCOLS, T), LDS tiled ----------------
__global__ __launch_bounds__(256)
void transpose_v_kernel(const float* __restrict__ v, __bf16* __restrict__ vT) {
    __shared__ float tile[32][33];
    const int tx = threadIdx.x;           // 0..31
    const int ty = threadIdx.y;           // 0..7
    const int col0 = blockIdx.x * 32;     // over VCOLS
    const int row0 = blockIdx.y * 32;     // over T
#pragma unroll
    for (int j = 0; j < 4; ++j)
        tile[ty + 8 * j][tx] = v[(size_t)(row0 + ty + 8 * j) * VCOLS + col0 + tx];
    __syncthreads();
#pragma unroll
    for (int j = 0; j < 4; ++j) {
        int d = col0 + ty + 8 * j;
        vT[(size_t)d * T_LEN + row0 + tx] = f2bf(tile[tx][ty + 8 * j]);
    }
}

// ---------------- flash attention: one wave per (head, 16-row q tile) ----------------
// q: bf16 (T, NHQ*DH) roped; k: bf16 (T, NKV*DH) roped; vT: bf16 (NKV*DH, T); o: bf16 (T, NHQ*DH)
__global__ __launch_bounds__(256)
void flash_attn_wmma(const __bf16* __restrict__ q, const __bf16* __restrict__ k,
                     const __bf16* __restrict__ vT, __bf16* __restrict__ o) {
    __shared__ __bf16 Pbuf[8 * 16 * 32];                  // per-wave 16x32 P tile
    const int lane   = threadIdx.x & 31;
    const int wave   = threadIdx.x >> 5;
    const int l16    = lane & 15;
    const int hiHalf = lane >> 4;

    const int wid = blockIdx.x * 8 + wave;                // 0 .. NHQ*(T/16)-1
    const int h   = wid & (NHQ - 1);
    const int qt  = wid >> 5;                             // q tile index
    const int t0  = qt * 16;
    const int kh  = h / GRP;

    // Q fragments (16 rows x 128 = 4 k-steps of 32)
    v16bf qf[4];
    const __bf16* qrow = q + ((size_t)(t0 + l16) * NHQ + h) * DH;
#pragma unroll
    for (int kk = 0; kk < 4; ++kk)
        qf[kk] = load_a_frag(qrow + kk * 32 + hiHalf * 8);

    const v8f zero = {0.f,0.f,0.f,0.f,0.f,0.f,0.f,0.f};
    v8f of[8];
#pragma unroll
    for (int nf = 0; nf < 8; ++nf) of[nf] = zero;
    float mrow[8], lrow[8];
#pragma unroll
    for (int r = 0; r < 8; ++r) { mrow[r] = -1e30f; lrow[r] = 0.f; }

    __bf16* P = Pbuf + wave * (16 * 32);
    const float scale = 0.08838834764831845f;             // 1/sqrt(128)

    for (int s0 = 0; s0 <= t0 + 15; s0 += 32) {
        // ---- S = Q K^T for keys [s0, s0+32) ----
        v8f S0 = zero, S1 = zero;
        const __bf16* k0p = k + ((size_t)(s0 + l16) * NKV + kh) * DH + hiHalf * 8;
        const __bf16* k1p = k0p + (size_t)16 * NKV * DH;
#pragma unroll
        for (int kk = 0; kk < 4; ++kk) {
            v16bf kb0 = load_a_frag(k0p + kk * 32);
            v16bf kb1 = load_a_frag(k1p + kk * 32);
            S0 = wmma_bf16(qf[kk], kb0, S0);
            S1 = wmma_bf16(qf[kk], kb1, S1);
        }
        // ---- online softmax (rows live across 16-lane halves) ----
        float p0[8], p1[8], alpha[8];
#pragma unroll
        for (int r = 0; r < 8; ++r) {
            int tg = t0 + r + hiHalf * 8;
            int sA = s0 + l16;
            float a = (sA      <= tg) ? S0[r] * scale : -1e30f;
            float b = (sA + 16 <= tg) ? S1[r] * scale : -1e30f;
            float tmax = fmaxf(a, b);
#pragma unroll
            for (int msk = 1; msk < 16; msk <<= 1)
                tmax = fmaxf(tmax, __shfl_xor(tmax, msk, 32));
            float nm = fmaxf(mrow[r], tmax);
            float al = __expf(mrow[r] - nm);
            float e0 = __expf(a - nm);
            float e1 = __expf(b - nm);
            float rs = e0 + e1;
#pragma unroll
            for (int msk = 1; msk < 16; msk <<= 1)
                rs += __shfl_xor(rs, msk, 32);
            lrow[r] = lrow[r] * al + rs;
            mrow[r] = nm;
            p0[r] = e0; p1[r] = e1; alpha[r] = al;
        }
#pragma unroll
        for (int nf = 0; nf < 8; ++nf)
#pragma unroll
            for (int r = 0; r < 8; ++r) of[nf][r] *= alpha[r];

        // ---- C-layout P -> LDS -> A-layout fragment ----
#pragma unroll
        for (int r = 0; r < 8; ++r) {
            int m = r + hiHalf * 8;
            P[m * 32 + l16]      = f2bf(p0[r]);
            P[m * 32 + 16 + l16] = f2bf(p1[r]);
        }
        v16bf pf = load_a_frag(P + l16 * 32 + hiHalf * 8);

        // ---- O += P V  (V^T rows are contiguous in s) ----
        const __bf16* vrow = vT + (size_t)(kh * DH) * T_LEN + s0 + hiHalf * 8;
#pragma unroll
        for (int nf = 0; nf < 8; ++nf) {
            int d = nf * 16 + l16;
            v16bf vf = load_a_frag(vrow + (size_t)d * T_LEN);
            of[nf] = wmma_bf16(pf, vf, of[nf]);
        }
    }
    // ---- normalize + store bf16 ----
#pragma unroll
    for (int r = 0; r < 8; ++r) lrow[r] = 1.0f / lrow[r];
#pragma unroll
    for (int nf = 0; nf < 8; ++nf) {
        int col = h * DH + nf * 16 + l16;
#pragma unroll
        for (int r = 0; r < 8; ++r) {
            int tg = t0 + r + hiHalf * 8;
            o[(size_t)tg * (NHQ * DH) + col] = f2bf(of[nf][r] * lrow[r]);
        }
    }
}

// ---------------- host orchestration ----------------
extern "C" void kernel_launch(void* const* d_in, const int* in_sizes, int n_in,
                              void* d_out, int out_size, void* d_ws, size_t ws_size,
                              hipStream_t stream) {
    (void)in_sizes; (void)n_in; (void)out_size; (void)ws_size;
    const float* x   = (const float*)d_in[0];
    const int*   pos = (const int*)d_in[1];
    const float* Wq  = (const float*)d_in[2];
    const float* Wk  = (const float*)d_in[3];
    const float* Wv  = (const float*)d_in[4];
    const float* Wo  = (const float*)d_in[5];
    float* out = (float*)d_out;

    char* ws = (char*)d_ws;
    size_t off = 0;
    auto alloc = [&](size_t bytes) -> void* {
        void* p = ws + off;
        off += (bytes + 255) & ~(size_t)255;
        return p;
    };
    __bf16* x_bf  = (__bf16*)alloc((size_t)T_LEN * K_IN * 2);
    __bf16* Wq_bf = (__bf16*)alloc((size_t)NHQ * DH * K_IN * 2);
    __bf16* Wk_bf = (__bf16*)alloc((size_t)NKV * DH * K_IN * 2);
    __bf16* Wv_bf = (__bf16*)alloc((size_t)NKV * DH * K_IN * 2);
    __bf16* Wo_bf = (__bf16*)alloc((size_t)HIDDEN * NHQ * DH * 2);
    float*  q_f   = (float*)alloc((size_t)T_LEN * NHQ * DH * 4);
    float*  k_f   = (float*)alloc((size_t)T_LEN * NKV * DH * 4);
    float*  v_f   = (float*)alloc((size_t)T_LEN * NKV * DH * 4);
    __bf16* q_bf  = (__bf16*)alloc((size_t)T_LEN * NHQ * DH * 2);
    __bf16* k_bf  = (__bf16*)alloc((size_t)T_LEN * NKV * DH * 2);
    __bf16* vT_bf = (__bf16*)alloc((size_t)NKV * DH * T_LEN * 2);
    __bf16* o_bf  = (__bf16*)alloc((size_t)T_LEN * NHQ * DH * 2);

    auto cvt = [&](const float* src, __bf16* dst, long n) {
        long n8 = n / 8;
        f32_to_bf16_vec8<<<dim3((unsigned)((n8 + 255) / 256)), dim3(256), 0, stream>>>(src, dst, n8);
    };
    cvt(x,  x_bf,  (long)T_LEN * K_IN);
    cvt(Wq, Wq_bf, (long)NHQ * DH * K_IN);
    cvt(Wk, Wk_bf, (long)NKV * DH * K_IN);
    cvt(Wv, Wv_bf, (long)NKV * DH * K_IN);
    cvt(Wo, Wo_bf, (long)HIDDEN * NHQ * DH);

    // QKV projections (block tile 128x256)
    gemm_bf16_wmma<<<dim3((NHQ * DH) / 256, T_LEN / 128), dim3(256), 0, stream>>>(
        x_bf, Wq_bf, q_f, T_LEN, NHQ * DH, K_IN);
    gemm_bf16_wmma<<<dim3((NKV * DH) / 256, T_LEN / 128), dim3(256), 0, stream>>>(
        x_bf, Wk_bf, k_f, T_LEN, NKV * DH, K_IN);
    gemm_bf16_wmma<<<dim3((NKV * DH) / 256, T_LEN / 128), dim3(256), 0, stream>>>(
        x_bf, Wv_bf, v_f, T_LEN, NKV * DH, K_IN);

    // RoPE on q, k
    rope_f32_to_bf16<<<dim3(T_LEN * NHQ), dim3(64), 0, stream>>>(q_f, q_bf, pos, NHQ);
    rope_f32_to_bf16<<<dim3(T_LEN * NKV), dim3(64), 0, stream>>>(k_f, k_bf, pos, NKV);

    // V -> V^T (bf16), tiled transpose
    transpose_v_kernel<<<dim3(VCOLS / 32, T_LEN / 32), dim3(32, 8), 0, stream>>>(v_f, vT_bf);

    // attention: NHQ * (T/16) waves, 8 waves per block
    flash_attn_wmma<<<dim3((NHQ * (T_LEN / 16)) / 8), dim3(256), 0, stream>>>(
        q_bf, k_bf, vT_bf, o_bf);

    // output projection -> f32 out
    gemm_bf16_wmma<<<dim3(HIDDEN / 256, T_LEN / 128), dim3(256), 0, stream>>>(
        o_bf, Wo_bf, out, T_LEN, HIDDEN, NHQ * DH);
}
